// MockMoEModel_6562710028723
// MI455X (gfx1250) — matrix operand order repaired and checked by reading, not compiled, not run on hardware
//
#include <hip/hip_runtime.h>
#include <cstdint>
#include <cstddef>

// ---------------------------------------------------------------------------
// Types for CDNA5 WMMA (wave32, 16x16x32 bf16 -> f32)
// ---------------------------------------------------------------------------
typedef __bf16 bf16;
typedef __attribute__((ext_vector_type(16))) __bf16 v16bf;
typedef __attribute__((ext_vector_type(8)))  __bf16 v8bf;
typedef __attribute__((ext_vector_type(4)))  __bf16 v4bf;
typedef __attribute__((ext_vector_type(8)))  float  v8f;

// Model dimensions (fixed by the reference)
constexpr int Bc = 4, Sc = 512, Hc = 1024, Ic = 4096, Ec = 8, Vc = 32000,
              Lc = 2, NHc = 8;
constexpr int Tc  = Bc * Sc;      // 2048 tokens
constexpr int DHc = Hc / NHc;     // 128 head dim

// ---------------------------------------------------------------------------
// Tiled WMMA GEMM:  C[M,N] (+)= A[M,K] * op(B) + bias
//   BT=true : B stored [N,K] (weight layout, C = A @ B^T)
//   BT=false: B stored [K,N]
//   MODE: 0 = direct rows; 1 = gather A rows via rowIdx (C rows compact);
//         2 = A rows compact, scatter C rows via rowIdx
//   Batched over blockIdx.z with 2-level strides: z = zb*zDiv + zh
//   Dynamic row bound from cntPtr (device-side M for MoE dispatch)
//
// Block = 256 threads = 8 waves; block tile 128x64; wave tile 32x32
// (2x2 of 16x16 WMMA tiles -> 4 v_wmma per 32-K step, 2x fragment reuse).
// LDS double-buffered: one barrier per K-step.
// ---------------------------------------------------------------------------
template<bool BT, bool ACCUM, int MODE>
__global__ __launch_bounds__(256) void gemm_wmma_kernel(
    const bf16* __restrict__ A, int lda, long long asb, long long ash,
    const bf16* __restrict__ Bm, int ldb, long long bsb, long long bsh,
    float* __restrict__ C, int ldc, long long csb, long long csh,
    const float* __restrict__ bias,
    int M, int N, int Kd, int zDiv,
    const int* __restrict__ rowIdx, const int* __restrict__ cntPtr)
{
    __shared__ bf16 As[2][128][40];   // +8 pad keeps 16B alignment
    __shared__ bf16 Bs[2][64][40];    // stored [n][k]

    const int z  = blockIdx.z;
    const int zb = z / zDiv, zh = z - zb * zDiv;
    A  += (size_t)(zb * asb + zh * ash);
    Bm += (size_t)(zb * bsb + zh * bsh);
    C  += (size_t)(zb * csb + zh * csh);

    const int Mv = cntPtr ? *cntPtr : M;   // dynamic row count (MoE)
    const int m0 = blockIdx.y * 128;
    const int n0 = blockIdx.x * 64;
    if (m0 >= Mv) return;

    const int tid  = threadIdx.x;
    const int lane = tid & 31;
    const int wave = tid >> 5;
    const int wm = (wave >> 1) << 5;       // 0,32,64,96
    const int wn = (wave & 1) << 5;        // 0,32

    // ---- loop-invariant staging coordinates ----
    // A tile: 128 rows x 32 k; each thread stages one row-half (16 bf16)
    const int arow = tid >> 1;             // 0..127
    const int acol = (tid & 1) << 4;       // 0 or 16
    int a_grow = -1;                       // resolved source row (gather once)
    {
        const int gm = m0 + arow;
        if (gm < Mv) a_grow = (MODE == 1) ? rowIdx[gm] : gm;
    }
    // B tile: 64 rows(n) x 32 k; each thread stages 8 bf16
    const int brow = tid >> 2;             // 0..63
    const int bci  = (tid & 3) << 3;       // 0,8,16,24
    const int b_gn = n0 + brow;

    auto loadA = [&](int kb, v8bf& a0, v8bf& a1) {
        a0 = {}; a1 = {};
        if (a_grow >= 0) {
            const bf16* p = A + (size_t)a_grow * lda + kb + acol;
            a0 = *(const v8bf*)p;
            a1 = *(const v8bf*)(p + 8);
            if (kb + 32 < Kd) __builtin_prefetch(p + 32, 0, 3);
        }
    };
    auto loadB = [&](int kb, v8bf& b0) {
        b0 = {};
        if (b_gn < N) {
            if (BT) {
                b0 = *(const v8bf*)(Bm + (size_t)b_gn * ldb + kb + bci);
            } else {
                #pragma unroll
                for (int j = 0; j < 8; ++j)
                    b0[j] = Bm[(size_t)(kb + bci + j) * ldb + b_gn];
            }
        }
    };
    auto storeA = [&](int buf, v8bf a0, v8bf a1) {
        *(v8bf*)&As[buf][arow][acol]     = a0;
        *(v8bf*)&As[buf][arow][acol + 8] = a1;
    };
    auto storeB = [&](int buf, v8bf b0) {
        *(v8bf*)&Bs[buf][brow][bci] = b0;
    };
    auto mk16 = [](v8bf lo, v8bf hi) {
        v16bf r;
        #pragma unroll
        for (int i = 0; i < 8; ++i) { r[i] = lo[i]; r[i + 8] = hi[i]; }
        return r;
    };

    v8f acc[2][2] = {};

    // ---- prologue: stage K-slab 0 ----
    v8bf ra0, ra1, rb0;
    loadA(0, ra0, ra1); loadB(0, rb0);
    storeA(0, ra0, ra1); storeB(0, rb0);
    __syncthreads();

    for (int kb = 0; kb < Kd; kb += 32) {
        const int  cur  = (kb >> 5) & 1;
        const bool more = (kb + 32) < Kd;
        if (more) { loadA(kb + 32, ra0, ra1); loadB(kb + 32, rb0); }

        // ---- fragments (ISA 7.12.2 layouts) ----
        const int ka = (lane >> 4) << 3;   // A: K chunks {ka..+7, ka+16..+23}
        const int r0 = wm + (lane & 15);
        v16bf af0 = mk16(*(const v8bf*)&As[cur][r0][ka],
                         *(const v8bf*)&As[cur][r0][ka + 16]);
        v16bf af1 = mk16(*(const v8bf*)&As[cur][r0 + 16][ka],
                         *(const v8bf*)&As[cur][r0 + 16][ka + 16]);

        const int kb2 = (lane >> 4) << 3;  // B: K chunks {kb2..+7, kb2+16..+23}
        const int c0  = wn + (lane & 15);
        v16bf bf0 = mk16(*(const v8bf*)&Bs[cur][c0][kb2],
                         *(const v8bf*)&Bs[cur][c0][kb2 + 16]);
        v16bf bf1 = mk16(*(const v8bf*)&Bs[cur][c0 + 16][kb2],
                         *(const v8bf*)&Bs[cur][c0 + 16][kb2 + 16]);

        acc[0][0] = __builtin_amdgcn_wmma_f32_16x16x32_bf16(
                        false, af0, false, bf0, (short)0, acc[0][0], false, false);
        acc[0][1] = __builtin_amdgcn_wmma_f32_16x16x32_bf16(
                        false, af0, false, bf1, (short)0, acc[0][1], false, false);
        acc[1][0] = __builtin_amdgcn_wmma_f32_16x16x32_bf16(
                        false, af1, false, bf0, (short)0, acc[1][0], false, false);
        acc[1][1] = __builtin_amdgcn_wmma_f32_16x16x32_bf16(
                        false, af1, false, bf1, (short)0, acc[1][1], false, false);

        if (more) { storeA(cur ^ 1, ra0, ra1); storeB(cur ^ 1, rb0); }
        __syncthreads();
    }

    // ---- epilogue: C layout VGPR v -> m = (lane>>4)*8 + v, n = lane&15 ----
    const int lm = (lane >> 4) << 3;
    const int ln = lane & 15;
    #pragma unroll
    for (int ti = 0; ti < 2; ++ti) {
        const int mbase = m0 + wm + ti * 16 + lm;
        #pragma unroll
        for (int tj = 0; tj < 2; ++tj) {
            const int n = n0 + wn + tj * 16 + ln;
            if (n >= N) continue;
            const float bv = bias ? bias[n] : 0.0f;
            #pragma unroll
            for (int v = 0; v < 8; ++v) {
                const int m = mbase + v;
                if (m < Mv) {
                    int orow = m;
                    if (MODE == 2) orow = rowIdx[m];
                    float* cp = C + (size_t)orow * ldc + n;
                    const float val = acc[ti][tj][v] + bv;
                    if (ACCUM) *cp += val; else *cp = val;
                }
            }
        }
    }
}

// ---------------------------------------------------------------------------
// Support kernels
// ---------------------------------------------------------------------------
__global__ __launch_bounds__(256) void embed_kernel(
    const int* __restrict__ ids, const float* __restrict__ emb,
    float* __restrict__ x, int H)
{
    const int t  = blockIdx.x;
    const int id = ids[t];
    const float* src = emb + (size_t)id * H;
    float* dst = x + (size_t)t * H;
    for (int i = threadIdx.x; i < H; i += 256) dst[i] = src[i];
}

__global__ __launch_bounds__(256) void cvt_bf16_kernel(
    const float* __restrict__ in, bf16* __restrict__ out, long long n)
{
    long long i = ((long long)blockIdx.x * 256 + threadIdx.x) * 4;
    if (i + 3 < n) {
        const float4 f = *(const float4*)(in + i);
        v4bf o;
        o[0] = (bf16)f.x; o[1] = (bf16)f.y; o[2] = (bf16)f.z; o[3] = (bf16)f.w;
        *(v4bf*)(out + i) = o;
    } else {
        for (; i < n; ++i) out[i] = (bf16)in[i];
    }
}

__global__ __launch_bounds__(256) void layernorm_bf16_kernel(
    const float* __restrict__ x, const float* __restrict__ g,
    const float* __restrict__ b, bf16* __restrict__ out, int H)
{
    const int t = blockIdx.x;
    const float* row = x + (size_t)t * H;
    __shared__ float red[256];
    const int tid = threadIdx.x;
    float s = 0.f;
    for (int i = tid; i < H; i += 256) s += row[i];
    red[tid] = s; __syncthreads();
    for (int w = 128; w > 0; w >>= 1) { if (tid < w) red[tid] += red[tid + w]; __syncthreads(); }
    const float mu = red[0] / H;
    __syncthreads();
    float v = 0.f;
    for (int i = tid; i < H; i += 256) { const float d = row[i] - mu; v += d * d; }
    red[tid] = v; __syncthreads();
    for (int w = 128; w > 0; w >>= 1) { if (tid < w) red[tid] += red[tid + w]; __syncthreads(); }
    const float rs = rsqrtf(red[0] / H + 1e-5f);
    bf16* orow = out + (size_t)t * H;
    for (int i = tid; i < H; i += 256)
        orow[i] = (bf16)((row[i] - mu) * rs * g[i] + b[i]);
}

__global__ __launch_bounds__(256) void softmax_bf16_kernel(
    const float* __restrict__ in, bf16* __restrict__ out, int S, float scale)
{
    const int r = blockIdx.x;
    const float* row = in + (size_t)r * S;
    bf16* orow = out + (size_t)r * S;
    __shared__ float red[256];
    const int tid = threadIdx.x;
    float mx = -3.4e38f;
    for (int i = tid; i < S; i += 256) mx = fmaxf(mx, row[i]);
    red[tid] = mx; __syncthreads();
    for (int w = 128; w > 0; w >>= 1) { if (tid < w) red[tid] = fmaxf(red[tid], red[tid + w]); __syncthreads(); }
    mx = red[0]; __syncthreads();
    float sum = 0.f;
    for (int i = tid; i < S; i += 256) sum += __expf((row[i] - mx) * scale);
    red[tid] = sum; __syncthreads();
    for (int w = 128; w > 0; w >>= 1) { if (tid < w) red[tid] += red[tid + w]; __syncthreads(); }
    const float inv = 1.0f / red[0];
    for (int i = tid; i < S; i += 256)
        orow[i] = (bf16)(__expf((row[i] - mx) * scale) * inv);
}

__global__ __launch_bounds__(256) void router_kernel(
    const bf16* __restrict__ xn, const float* __restrict__ rw,
    const float* __restrict__ rb, float* __restrict__ logits, int H, int E)
{
    const int t = blockIdx.x;
    __shared__ float red[256];
    const int tid = threadIdx.x;
    const bf16* xr = xn + (size_t)t * H;
    for (int e = 0; e < E; ++e) {
        const float* wr = rw + (size_t)e * H;
        float s = 0.f;
        for (int i = tid; i < H; i += 256) s += (float)xr[i] * wr[i];
        red[tid] = s; __syncthreads();
        for (int w = 128; w > 0; w >>= 1) { if (tid < w) red[tid] += red[tid + w]; __syncthreads(); }
        if (tid == 0) logits[(size_t)t * E + e] = red[0] + rb[e];
        __syncthreads();
    }
}

// top-2 selection + per-expert token-list dispatch (order in list is
// atomics-nondeterministic but the final scatter-add result is not)
__global__ __launch_bounds__(256) void topk_dispatch_kernel(
    const float* __restrict__ logits, int* __restrict__ lists,
    int* __restrict__ cnts, int T, int E)
{
    const int t = blockIdx.x * blockDim.x + threadIdx.x;
    if (t >= T) return;
    const float* l = logits + (size_t)t * E;
    int i0 = 0; float v0 = l[0];
    for (int e = 1; e < E; ++e) if (l[e] > v0) { v0 = l[e]; i0 = e; }
    int i1 = -1; float v1 = -3.4e38f;
    for (int e = 0; e < E; ++e) if (e != i0 && l[e] > v1) { v1 = l[e]; i1 = e; }
    const int p0 = atomicAdd(&cnts[i0], 1); lists[(size_t)i0 * T + p0] = t;
    const int p1 = atomicAdd(&cnts[i1], 1); lists[(size_t)i1 * T + p1] = t;
}

__global__ __launch_bounds__(256) void gelu_bf16_kernel(
    const float* __restrict__ h, bf16* __restrict__ out,
    const int* __restrict__ cnt, int I)
{
    const long long idx = (long long)blockIdx.x * 256 + threadIdx.x;
    const int row = (int)(idx / I);
    if (row >= *cnt) return;
    const float v = h[idx];
    out[idx] = (bf16)(0.5f * v * (1.0f + erff(v * 0.70710678118f)));
}

__global__ void copy_counts_kernel(const int* __restrict__ c,
                                   float* __restrict__ cf, int E)
{
    const int e = threadIdx.x;
    if (e < E) cf[e] = (float)c[e];
}

__global__ void aux_kernel(const float* __restrict__ cf, float* __restrict__ out,
                           int L, int E, float target)
{
    float total = 0.f;
    for (int l = 0; l < L; ++l) {
        float s = 0.f;
        for (int e = 0; e < E; ++e) { const float d = cf[l * E + e] - target; s += d * d; }
        total += (s / E) * 0.01f;
    }
    *out = total;
}

// ---------------------------------------------------------------------------
// Host launcher
// ---------------------------------------------------------------------------
template<bool BT, bool ACCUM, int MODE>
static void run_gemm(hipStream_t s,
                     const bf16* A, int lda, long long asb, long long ash,
                     const bf16* Bm, int ldb, long long bsb, long long bsh,
                     float* C, int ldc, long long csb, long long csh,
                     const float* bias, int M, int N, int Kd,
                     int zDiv, int batches,
                     const int* rowIdx, const int* cnt)
{
    dim3 grid((unsigned)((N + 63) / 64), (unsigned)((M + 127) / 128), (unsigned)batches);
    gemm_wmma_kernel<BT, ACCUM, MODE><<<grid, 256, 0, s>>>(
        A, lda, asb, ash, Bm, ldb, bsb, bsh, C, ldc, csb, csh,
        bias, M, N, Kd, zDiv, rowIdx, cnt);
}

extern "C" void kernel_launch(void* const* d_in, const int* in_sizes, int n_in,
                              void* d_out, int out_size, void* d_ws, size_t ws_size,
                              hipStream_t stream)
{
    (void)in_sizes; (void)n_in; (void)out_size; (void)ws_size;

    const int*   ids    = (const int*)  d_in[0];
    const float* emb    = (const float*)d_in[1];
    const float* ln1_g  = (const float*)d_in[2];
    const float* ln1_b  = (const float*)d_in[3];
    const float* ain_w  = (const float*)d_in[4];
    const float* ain_b  = (const float*)d_in[5];
    const float* aout_w = (const float*)d_in[6];
    const float* aout_b = (const float*)d_in[7];
    const float* ln2_g  = (const float*)d_in[8];
    const float* ln2_b  = (const float*)d_in[9];
    const float* rw     = (const float*)d_in[10];
    const float* rb     = (const float*)d_in[11];
    const float* f1w    = (const float*)d_in[12];
    const float* f1b    = (const float*)d_in[13];
    const float* f2w    = (const float*)d_in[14];
    const float* f2b    = (const float*)d_in[15];
    const float* lm_w   = (const float*)d_in[16];
    const float* lm_b   = (const float*)d_in[17];
    float* out = (float*)d_out;

    // Workspace layout (~230 MB), 256B aligned slices
    char* base = (char*)d_ws;
    size_t off = 0;
    auto alloc = [&](size_t bytes) -> char* {
        char* p = base + off;
        off += (bytes + 255) & ~(size_t)255;
        return p;
    };
    float* x        = (float*)alloc((size_t)Tc * Hc * 4);
    float* qkv      = (float*)alloc((size_t)Tc * 3 * Hc * 4);
    bf16*  qkv_bf   = (bf16*) alloc((size_t)Tc * 3 * Hc * 2);
    bf16*  xn_bf    = (bf16*) alloc((size_t)Tc * Hc * 2);
    bf16*  act_bf   = (bf16*) alloc((size_t)Tc * Hc * 2);
    float* scoresB  = (float*)alloc((size_t)Bc * NHc * Sc * Sc * 4);
    bf16*  probs_bf = (bf16*) alloc((size_t)Bc * NHc * Sc * Sc * 2);
    float* attn_cat = (float*)alloc((size_t)Tc * Hc * 4);
    float* hbuf     = (float*)alloc((size_t)Tc * Ic * 4);
    bf16*  h_bf     = (bf16*) alloc((size_t)Tc * Ic * 2);
    float* rlog     = (float*)alloc((size_t)Tc * Ec * 4);
    int*   lists    = (int*)  alloc((size_t)Ec * Tc * 4);
    int*   cnts     = (int*)  alloc((size_t)Ec * 4);
    float* countsf  = (float*)alloc((size_t)Lc * Ec * 4);
    bf16*  wA       = (bf16*) alloc((size_t)Vc * Hc * 2);   // big weight scratch
    bf16*  wB       = (bf16*) alloc((size_t)Ic * Hc * 2);   // second weight scratch

    auto cvt = [&](const float* src, bf16* dst, long long n) {
        cvt_bf16_kernel<<<(unsigned)((n / 4 + 255) / 256), 256, 0, stream>>>(src, dst, n);
    };

    // x = embedding[ids]
    embed_kernel<<<Tc, 256, 0, stream>>>(ids, emb, x, Hc);

    const float scale = 0.08838834764831845f;   // 1/sqrt(DH=128)

    for (int l = 0; l < Lc; ++l) {
        // ---- LN1 -> bf16 ----
        layernorm_bf16_kernel<<<Tc, 256, 0, stream>>>(
            x, ln1_g + (size_t)l * Hc, ln1_b + (size_t)l * Hc, xn_bf, Hc);

        // ---- QKV projection ----
        cvt(ain_w + (size_t)l * 3 * Hc * Hc, wA, (long long)3 * Hc * Hc);
        run_gemm<true, false, 0>(stream, xn_bf, Hc, 0, 0, wA, Hc, 0, 0,
                                 qkv, 3 * Hc, 0, 0,
                                 ain_b + (size_t)l * 3 * Hc,
                                 Tc, 3 * Hc, Hc, 1, 1, nullptr, nullptr);
        cvt(qkv, qkv_bf, (long long)Tc * 3 * Hc);

        // ---- scores = Q @ K^T, batched over (b, head) ----
        run_gemm<true, false, 0>(stream,
            qkv_bf,            3 * Hc, (long long)Sc * 3 * Hc, DHc,
            qkv_bf + Hc,       3 * Hc, (long long)Sc * 3 * Hc, DHc,
            scoresB,           Sc,     (long long)NHc * Sc * Sc, (long long)Sc * Sc,
            nullptr, Sc, Sc, DHc, NHc, Bc * NHc, nullptr, nullptr);

        softmax_bf16_kernel<<<Bc * NHc * Sc, 256, 0, stream>>>(scoresB, probs_bf, Sc, scale);

        // ---- O = P @ V, writes directly in [B,S,H] concat layout ----
        run_gemm<false, false, 0>(stream,
            probs_bf,          Sc,     (long long)NHc * Sc * Sc, (long long)Sc * Sc,
            qkv_bf + 2 * Hc,   3 * Hc, (long long)Sc * 3 * Hc, DHc,
            attn_cat,          Hc,     (long long)Sc * Hc, DHc,
            nullptr, Sc, DHc, Sc, NHc, Bc * NHc, nullptr, nullptr);

        // ---- out-proj, accumulated straight into residual x ----
        cvt(attn_cat, act_bf, (long long)Tc * Hc);
        cvt(aout_w + (size_t)l * Hc * Hc, wB, (long long)Hc * Hc);
        run_gemm<true, true, 0>(stream, act_bf, Hc, 0, 0, wB, Hc, 0, 0,
                                x, Hc, 0, 0, aout_b + (size_t)l * Hc,
                                Tc, Hc, Hc, 1, 1, nullptr, nullptr);

        // ---- LN2 -> bf16 ----
        layernorm_bf16_kernel<<<Tc, 256, 0, stream>>>(
            x, ln2_g + (size_t)l * Hc, ln2_b + (size_t)l * Hc, xn_bf, Hc);

        // ---- router + top-2 dispatch ----
        router_kernel<<<Tc, 256, 0, stream>>>(
            xn_bf, rw + (size_t)l * Ec * Hc, rb + (size_t)l * Ec, rlog, Hc, Ec);
        hipMemsetAsync(cnts, 0, Ec * sizeof(int), stream);
        topk_dispatch_kernel<<<(Tc + 255) / 256, 256, 0, stream>>>(rlog, lists, cnts, Tc, Ec);
        copy_counts_kernel<<<1, Ec, 0, stream>>>(cnts, countsf + (size_t)l * Ec, Ec);

        // ---- experts: gathered fc1 -> gelu -> fc2 scatter-add into x ----
        for (int e = 0; e < Ec; ++e) {
            const size_t we = (size_t)l * Ec + e;
            cvt(f1w + we * (size_t)Ic * Hc, wA, (long long)Ic * Hc);
            run_gemm<true, false, 1>(stream, xn_bf, Hc, 0, 0, wA, Hc, 0, 0,
                                     hbuf, Ic, 0, 0, f1b + we * (size_t)Ic,
                                     Tc, Ic, Hc, 1, 1,
                                     lists + (size_t)e * Tc, cnts + e);
            gelu_bf16_kernel<<<(unsigned)(((long long)Tc * Ic) / 256), 256, 0, stream>>>(
                hbuf, h_bf, cnts + e, Ic);
            cvt(f2w + we * (size_t)Hc * Ic, wB, (long long)Hc * Ic);
            run_gemm<true, true, 2>(stream, h_bf, Ic, 0, 0, wB, Ic, 0, 0,
                                    x, Hc, 0, 0, f2b + we * (size_t)Hc,
                                    Tc, Hc, Ic, 1, 1,
                                    lists + (size_t)e * Tc, cnts + e);
        }
    }

    // ---- LM head into d_out ----
    cvt(x, act_bf, (long long)Tc * Hc);
    cvt(lm_w, wA, (long long)Vc * Hc);
    run_gemm<true, false, 0>(stream, act_bf, Hc, 0, 0, wA, Hc, 0, 0,
                             out, Vc, 0, 0, lm_b,
                             Tc, Vc, Hc, 1, 1, nullptr, nullptr);

    // ---- aux loss scalar appended after the logits ----
    aux_kernel<<<1, 1, 0, stream>>>(countsf, out + (size_t)Tc * Vc,
                                    Lc, Ec, (float)Tc / Ec);
}
